// EncoderGNN_25202868093638
// MI455X (gfx1250) — matrix operand order repaired and checked by reading, not compiled, not run on hardware
//
#include <hip/hip_runtime.h>

// ---------------- CDNA5 / gfx1250 EncoderGNN ----------------
// Fused WMMA edge-MLP message passing + atomic scatter aggregation.
// wave32, v_wmma_f32_16x16x32_bf16 for all GEMMs.
// Edge kernel: 64-edge M-tiles (4x B-fragment reuse), A fragments loaded
// as uint4 so LDS traffic is ds_load_b128 (not 2addr_b32 pairs).

typedef unsigned int uint32;
typedef __attribute__((ext_vector_type(16))) __bf16 v16bf;
typedef __attribute__((ext_vector_type(8)))  float  v8f;

#define SDIM     256
#define VDIM     64
#define MSG_OUT  384
#define RBF_DIM  32
#define GRAPH    20
#define CUTOFF   7.5f
#define NLAYERS  5
#define EDGE_IN  545
#define KC1      18      // ceil(576/32) K-chunks for GEMM1
#define NT1      16      // 256/16 N-tiles for GEMM1
#define KC2      8       // 256/32 K-chunks for GEMM2
#define NT2      24      // 384/16 N-tiles for GEMM2
#define GEOW     17      // geo words (34 bf16: 32 rbf + a + pad) per edge
#define LNEPS    1e-5f
#define EPB      64      // edges per block in fused edge kernel
#define MT       4       // M-tiles (of 16 edges) per block
#define EDGE_SMEM (EPB * 288 * 4 + EPB * 256 * 2)   // Atile + Htile = 106,496 B

union Frag {
  v16bf v;
  uint32 u[8];
  uint4  q[2];
};

__device__ inline uint32 pack_bf2(float a, float b) {
  unsigned short ua = __builtin_bit_cast(unsigned short, (__bf16)a);
  unsigned short ub = __builtin_bit_cast(unsigned short, (__bf16)b);
  return (uint32)ua | ((uint32)ub << 16);
}
__device__ inline void atomAddF(float* p, float v) {
  __hip_atomic_fetch_add(p, v, __ATOMIC_RELAXED, __HIP_MEMORY_SCOPE_AGENT);
}
__device__ inline float waveSum(float v) {
#pragma unroll
  for (int o = 16; o; o >>= 1) v += __shfl_xor(v, o, 32);
  return v;
}
__device__ inline float waveMax(float v) {
#pragma unroll
  for (int o = 16; o; o >>= 1) v = fmaxf(v, __shfl_xor(v, o, 32));
  return v;
}
__device__ inline float siluf(float x) { return x / (1.0f + __expf(-x)); }

// A fragment (16-bit A layout) = two 16B-aligned uint4 groups:
// words row*strideW + 16*kc + kh*4 + {0..3} and + {8..11}
__device__ inline void loadAfrag(Frag& a, const uint32* base, int rowWord, int k0half, int kh) {
  const uint4* sp = (const uint4*)(base + rowWord + k0half + kh * 4);
  a.q[0] = sp[0];
  a.q[1] = sp[2];
}

// ---- center positions per graph (scatter_mean, 20 nodes/graph) ----
__global__ void k_center(const float* __restrict__ pos, float* __restrict__ posc, int G) {
  int g = blockIdx.x * blockDim.x + threadIdx.x;
  if (g >= G) return;
#pragma unroll
  for (int d = 0; d < 3; ++d) {
    float m = 0.f;
    for (int k = 0; k < GRAPH; ++k) m += pos[(size_t)(g * GRAPH + k) * 3 + d];
    m *= (1.0f / GRAPH);
    for (int k = 0; k < GRAPH; ++k)
      posc[(size_t)(g * GRAPH + k) * 3 + d] = pos[(size_t)(g * GRAPH + k) * 3 + d] - m;
  }
}

// ---- concat local+global edge lists into flat src/dst ----
__global__ void k_copy_edges(const int* __restrict__ loc, const int* __restrict__ glo,
                             int EL, int EG, int* __restrict__ srcA, int* __restrict__ dstA) {
  int e = blockIdx.x * blockDim.x + threadIdx.x;
  int E = EL + EG;
  if (e >= E) return;
  if (e < EL) { srcA[e] = loc[e]; dstA[e] = loc[EL + e]; }
  else        { int i = e - EL; srcA[e] = glo[i]; dstA[e] = glo[EG + i]; }
}

// ---- atom embedding: s = x @ W_atom + b (K=16, scalar is fine) ----
__global__ void k_embed(const float* __restrict__ x, const float* __restrict__ W,
                        const float* __restrict__ b, float* __restrict__ s,
                        __bf16* __restrict__ sbf, int N) {
  int idx = blockIdx.x * blockDim.x + threadIdx.x;
  if (idx >= N * SDIM) return;
  int n = idx >> 8, c = idx & 255;
  float acc = b[c];
#pragma unroll
  for (int k = 0; k < 16; ++k) acc += x[n * 16 + k] * W[k * SDIM + c];
  s[idx] = acc;
  sbf[idx] = (__bf16)acc;
}

// ---- edge geometry: rn, rbf||a (bf16 packed), in-degree counts ----
__global__ void k_geom(const int* __restrict__ srcA, const int* __restrict__ dstA,
                       const float* __restrict__ posc, float* __restrict__ rn,
                       uint32* __restrict__ geobf, int* __restrict__ counts, int E) {
  int e = blockIdx.x * blockDim.x + threadIdx.x;
  if (e >= E) return;
  int s_ = srcA[e], d_ = dstA[e];
  float ps0 = posc[s_ * 3 + 0], ps1 = posc[s_ * 3 + 1], ps2 = posc[s_ * 3 + 2];
  float pd0 = posc[d_ * 3 + 0], pd1 = posc[d_ * 3 + 1], pd2 = posc[d_ * 3 + 2];
  float r0 = pd0 - ps0, r1 = pd1 - ps1, r2 = pd2 - ps2;
  float a = pd0 * ps0 + pd1 * ps1 + pd2 * ps2;
  float dd = fmaxf(r0 * r0 + r1 * r1 + r2 * r2, 1e-6f);
  float d = sqrtf(dd);
  float invd = 1.0f / d;
  rn[(size_t)e * 3 + 0] = r0 * invd;
  rn[(size_t)e * 3 + 1] = r1 * invd;
  rn[(size_t)e * 3 + 2] = r2 * invd;
  const float step  = CUTOFF / (RBF_DIM - 1);
  const float gamma = 1.0f / (2.0f * (CUTOFF / RBF_DIM) * (CUTOFF / RBF_DIM));
  float rb[RBF_DIM];
#pragma unroll
  for (int j = 0; j < RBF_DIM; ++j) {
    float t = d - step * j;
    rb[j] = __expf(-gamma * t * t);
  }
#pragma unroll
  for (int w = 0; w < 16; ++w) geobf[(size_t)e * GEOW + w] = pack_bf2(rb[2 * w], rb[2 * w + 1]);
  geobf[(size_t)e * GEOW + 16] = pack_bf2(a, 0.0f);
  atomicAdd(&counts[d_], 1);
}

// ---- weight conversion: f32 [L,K,N] -> bf16, WMMA-B-fragment swizzled ----
// out word idx = ((l*KC + kc)*NT + nt)*256 + lane*8 + r
// B 16-bit 32x16 layout: lanes 0-15 K=0..7,16..23; lanes 16-31 K=8..15,24..31 (VGPR r: K=(r>>2)*16+8h+2(r&3),+1)
__global__ void k_swizzle(const float* __restrict__ W, uint32* __restrict__ out,
                          int K, int Ncols, int KC, int NT, int nL) {
  long idx = (long)blockIdx.x * blockDim.x + threadIdx.x;
  long total = (long)nL * KC * NT * 256;
  if (idx >= total) return;
  int r = (int)(idx & 7);
  int lane = (int)((idx >> 3) & 31);
  long t = idx >> 8;
  int nt = (int)(t % NT); t /= NT;
  int kc = (int)(t % KC);
  int l  = (int)(t / KC);
  int n = nt * 16 + (lane & 15);
  int k0 = kc * 32 + (r >> 2) * 16 + (lane >> 4) * 8 + (r & 3) * 2;
  const float* Wl = W + (long)l * K * Ncols;
  float a = (k0     < K) ? Wl[(long)k0 * Ncols + n]       : 0.f;
  float b = (k0 + 1 < K) ? Wl[(long)(k0 + 1) * Ncols + n] : 0.f;
  out[idx] = pack_bf2(a, b);
}

// ---- fused edge MLP: m_in->GEMM1->SiLU->GEMM2->atomic scatter ----
// 64 edges (4 M-tiles) per block; each wave: 4 M-tiles x 4 N-tiles (GEMM1),
// 4 M-tiles x 6 N-tiles in two groups of 3 (GEMM2). B fragments reused 4x.
__global__ void __launch_bounds__(128)
k_edge_fused(const uint32* __restrict__ sbf_u, const uint32* __restrict__ geobf_u,
             const int* __restrict__ srcA, const int* __restrict__ dstA,
             const float* __restrict__ rn, const float* __restrict__ vbuf,
             const uint32* __restrict__ w1sw, const float* __restrict__ bm1,
             const uint32* __restrict__ w2sw, const float* __restrict__ bm2,
             float* __restrict__ ms_agg, float* __restrict__ mv_agg, int E) {
  extern __shared__ char dsm[];
  uint32* Atile = (uint32*)dsm;                         // EPB x 288 u32 (m_in bf16)
  __bf16* Htile = (__bf16*)(dsm + EPB * 288 * 4);       // EPB x 256 bf16 (hidden)
  __shared__ int sSrc[EPB];
  __shared__ int sDst[EPB];
  const int tid = threadIdx.x;
  const int wave = tid >> 5, lane = tid & 31;
  const int e0 = blockIdx.x * EPB;
  if (tid < EPB) {
    int e = min(e0 + tid, E - 1);
    sSrc[tid] = srcA[e];
    sDst[tid] = dstA[e];
  }
  __syncthreads();
  // stage m_in = [s_src | s_dst | rbf,a,0-pad]
  for (int w = tid; w < EPB * 288; w += 128) {
    int row = w / 288, c2 = w - row * 288;
    uint32 val = 0u;
    if (c2 < 128)              val = sbf_u[(size_t)sSrc[row] * 128 + c2];
    else if (c2 < 256)         val = sbf_u[(size_t)sDst[row] * 128 + (c2 - 128)];
    else if (c2 < 256 + GEOW)  val = geobf_u[(size_t)min(e0 + row, E - 1) * GEOW + (c2 - 256)];
    Atile[w] = val;
  }
  __syncthreads();
  const int lm = lane & 15, kh = lane >> 4;

  // GEMM1: [64,576] x [576,256]; wave owns N-tiles [wave*4, wave*4+4)
  v8f acc[MT][4] = {};
  for (int kc = 0; kc < KC1; ++kc) {
    Frag a[MT];
    int k0half = kc * 16;   // (kc*32)>>1 words
#pragma unroll
    for (int mt = 0; mt < MT; ++mt)
      loadAfrag(a[mt], Atile, (mt * 16 + lm) * 288, k0half, kh);
#pragma unroll
    for (int t = 0; t < 4; ++t) {
      Frag b;
      const uint4* bp = (const uint4*)(w1sw + (((size_t)kc * NT1 + (wave * 4 + t)) * 32 + lane) * 8);
      b.q[0] = bp[0];
      b.q[1] = bp[1];
#pragma unroll
      for (int mt = 0; mt < MT; ++mt)
        acc[mt][t] = __builtin_amdgcn_wmma_f32_16x16x32_bf16(false, a[mt].v, false, b.v,
                                                             (short)0, acc[mt][t], false, false);
    }
  }
  // bias + SiLU -> bf16 hidden in LDS
#pragma unroll
  for (int t = 0; t < 4; ++t) {
    int col = (wave * 4 + t) * 16 + lm;
    float bias = bm1[col];
#pragma unroll
    for (int mt = 0; mt < MT; ++mt)
#pragma unroll
      for (int r = 0; r < 8; ++r) {
        int row = mt * 16 + r + 8 * kh;
        Htile[row * 256 + col] = (__bf16)siluf(acc[mt][t][r] + bias);
      }
  }
  __syncthreads();

  // GEMM2: [64,256] x [256,384]; wave owns N-tiles [wave*6, wave*6+6) in 2 groups
  const uint32* Hu = (const uint32*)Htile;
  for (int grp = 0; grp < 2; ++grp) {
    v8f acc2[MT][3] = {};
    for (int kc = 0; kc < KC2; ++kc) {
      Frag a[MT];
      int k0half = kc * 16;
#pragma unroll
      for (int mt = 0; mt < MT; ++mt)
        loadAfrag(a[mt], Hu, (mt * 16 + lm) * 128, k0half, kh);
#pragma unroll
      for (int t = 0; t < 3; ++t) {
        int nt = wave * 6 + grp * 3 + t;
        Frag b;
        const uint4* bp = (const uint4*)(w2sw + (((size_t)kc * NT2 + nt) * 32 + lane) * 8);
        b.q[0] = bp[0];
        b.q[1] = bp[1];
#pragma unroll
        for (int mt = 0; mt < MT; ++mt)
          acc2[mt][t] = __builtin_amdgcn_wmma_f32_16x16x32_bf16(false, a[mt].v, false, b.v,
                                                                (short)0, acc2[mt][t], false, false);
      }
    }
    // epilogue: bias + scatter (ms -> ms_agg; gv1*v[src], gv2*rn -> mv_agg)
#pragma unroll
    for (int t = 0; t < 3; ++t) {
      int n = (wave * 6 + grp * 3 + t) * 16 + lm;
      float bias = bm2[n];
#pragma unroll
      for (int mt = 0; mt < MT; ++mt)
#pragma unroll
        for (int r = 0; r < 8; ++r) {
          int row = mt * 16 + r + 8 * kh;
          int e = e0 + row;
          if (e >= E) continue;
          float val = acc2[mt][t][r] + bias;
          int dn = sDst[row];
          if (n < SDIM) {
            atomAddF(&ms_agg[(size_t)dn * SDIM + n], val);
          } else if (n < SDIM + VDIM) {
            int j = n - SDIM, sn = sSrc[row];
#pragma unroll
            for (int d = 0; d < 3; ++d)
              atomAddF(&mv_agg[((size_t)dn * 3 + d) * VDIM + j],
                       val * vbuf[((size_t)sn * 3 + d) * VDIM + j]);
          } else {
            int j = n - SDIM - VDIM;
#pragma unroll
            for (int d = 0; d < 3; ++d)
              atomAddF(&mv_agg[((size_t)dn * 3 + d) * VDIM + j], val * rn[(size_t)e * 3 + d]);
          }
        }
    }
  }
}

// ---- node update: s = LN(s + ms/deg), v += mv/deg; one wave per node ----
__global__ void __launch_bounds__(256)
k_node_update(float* __restrict__ s, __bf16* __restrict__ sbf, float* __restrict__ vbuf,
              const float* __restrict__ ms, const float* __restrict__ mv,
              const int* __restrict__ counts, const float* __restrict__ lng,
              const float* __restrict__ lnb, int N) {
  int wave = threadIdx.x >> 5, lane = threadIdx.x & 31;
  int n = blockIdx.x * 8 + wave;
  if (n >= N) return;
  float deg = fmaxf((float)counts[n], 1.0f);
  float inv_deg = 1.0f / deg;
  float x[8];
  float loc = 0.f;
#pragma unroll
  for (int i = 0; i < 8; ++i) {
    int c = i * 32 + lane;
    x[i] = s[(size_t)n * SDIM + c] + ms[(size_t)n * SDIM + c] * inv_deg;
    loc += x[i];
  }
  float mu = waveSum(loc) * (1.0f / SDIM);
  float vs = 0.f;
#pragma unroll
  for (int i = 0; i < 8; ++i) { float d = x[i] - mu; vs += d * d; }
  float inv = rsqrtf(waveSum(vs) * (1.0f / SDIM) + LNEPS);
#pragma unroll
  for (int i = 0; i < 8; ++i) {
    int c = i * 32 + lane;
    float y = (x[i] - mu) * inv * lng[c] + lnb[c];
    s[(size_t)n * SDIM + c] = y;
    sbf[(size_t)n * SDIM + c] = (__bf16)y;
  }
#pragma unroll
  for (int i = 0; i < 6; ++i) {
    int c = i * 32 + lane;
    vbuf[(size_t)n * 192 + c] += mv[(size_t)n * 192 + c] * inv_deg;
  }
}

// ---- generic node GEMM: out = act(Abf[M,256] @ Bsw + bias), WMMA ----
__global__ void __launch_bounds__(128)
k_gemm_node(const uint32* __restrict__ Abf_u, const uint32* __restrict__ Bsw,
            const float* __restrict__ bias, float* __restrict__ outF,
            __bf16* __restrict__ outBF, int doSilu, int M) {
  const int tid = threadIdx.x, wave = tid >> 5, lane = tid & 31;
  const int m = lane & 15, kh = lane >> 4;
  const int row0 = blockIdx.x * 16;
  v8f acc[4] = {};
  for (int kc = 0; kc < 8; ++kc) {
    Frag a;
    int row = min(row0 + m, M - 1);
    loadAfrag(a, Abf_u, row * 128, kc * 16, kh);
#pragma unroll
    for (int t = 0; t < 4; ++t) {
      Frag b;
      const uint4* bp = (const uint4*)(Bsw + (((size_t)kc * 16 + (wave * 4 + t)) * 32 + lane) * 8);
      b.q[0] = bp[0];
      b.q[1] = bp[1];
      acc[t] = __builtin_amdgcn_wmma_f32_16x16x32_bf16(false, a.v, false, b.v,
                                                       (short)0, acc[t], false, false);
    }
  }
#pragma unroll
  for (int t = 0; t < 4; ++t) {
    int n = wave * 64 + t * 16 + m;
    float bv = bias[n];
#pragma unroll
    for (int r = 0; r < 8; ++r) {
      int row = row0 + r + 8 * kh;
      if (row >= M) continue;
      float val = acc[t][r] + bv;
      if (doSilu) val = siluf(val);
      if (outF)  outF[(size_t)row * 256 + n] = val;
      if (outBF) outBF[(size_t)row * 256 + n] = (__bf16)val;
    }
  }
}

// ---- gate head: gate[n] = hg[n,:] . Wg2 + bg2 ----
__global__ void k_gate(const __bf16* __restrict__ hg, const float* __restrict__ Wg2,
                       const float* __restrict__ bg2, float* __restrict__ gate, int N) {
  int n = blockIdx.x * blockDim.x + threadIdx.x;
  if (n >= N) return;
  float acc = bg2[0];
  for (int k = 0; k < 256; ++k) acc += (float)hg[(size_t)n * 256 + k] * Wg2[k];
  gate[n] = acc;
}

// ---- softmax-attention pooling: one wave per graph ----
__global__ void __launch_bounds__(256)
k_pool(const float* __restrict__ gate, const float* __restrict__ nodef,
       float* __restrict__ out, int G) {
  int wave = threadIdx.x >> 5, lane = threadIdx.x & 31;
  int g = blockIdx.x * 8 + wave;
  if (g >= G) return;
  float gv = (lane < GRAPH) ? gate[g * GRAPH + lane] : -3.0e38f;
  float gmax = waveMax(gv);
  float ex = (lane < GRAPH) ? __expf(gv - gmax) : 0.f;
  float denom = waveSum(ex) + 1e-16f;
  float invden = 1.0f / denom;
#pragma unroll
  for (int i = 0; i < 8; ++i) {
    int c = i * 32 + lane;
    float acc = 0.f;
    for (int k = 0; k < GRAPH; ++k) {
      float w = __expf(gate[g * GRAPH + k] - gmax);
      acc += w * nodef[(size_t)(g * GRAPH + k) * 256 + c];
    }
    out[(size_t)g * 256 + c] = acc * invden;
  }
}

extern "C" void kernel_launch(void* const* d_in, const int* in_sizes, int n_in,
                              void* d_out, int out_size, void* d_ws, size_t ws_size,
                              hipStream_t stream) {
  (void)n_in; (void)out_size; (void)ws_size;
  const float* x      = (const float*)d_in[0];
  const float* pos    = (const float*)d_in[1];
  const float* W_atom = (const float*)d_in[2];
  const float* b_atom = (const float*)d_in[3];
  const float* Wm1    = (const float*)d_in[4];
  const float* bm1    = (const float*)d_in[5];
  const float* Wm2    = (const float*)d_in[6];
  const float* bm2    = (const float*)d_in[7];
  const float* ln_g   = (const float*)d_in[8];
  const float* ln_b   = (const float*)d_in[9];
  const float* W_lat  = (const float*)d_in[10];
  const float* b_lat  = (const float*)d_in[11];
  const float* Wn1    = (const float*)d_in[12];
  const float* bn1    = (const float*)d_in[13];
  const float* Wn2    = (const float*)d_in[14];
  const float* bn2    = (const float*)d_in[15];
  const float* Wg1    = (const float*)d_in[16];
  const float* bg1    = (const float*)d_in[17];
  const float* Wg2    = (const float*)d_in[18];
  const float* bg2    = (const float*)d_in[19];
  const int* eloc     = (const int*)d_in[20];
  const int* eglo     = (const int*)d_in[21];

  const int N  = in_sizes[0] / 16;      // 20000 nodes
  const int G  = N / GRAPH;             // 1000 graphs
  const int EL = in_sizes[20] / 2;
  const int EG = in_sizes[21] / 2;
  const int E  = EL + EG;

  // bump allocator over d_ws
  char* base = (char*)d_ws;
  size_t off = 0;
  auto alloc = [&](size_t bytes) -> void* {
    off = (off + 255) & ~(size_t)255;
    void* p = base + off;
    off += bytes;
    return p;
  };
  float*  posc   = (float*)alloc((size_t)N * 3 * 4);
  int*    srcA   = (int*)alloc((size_t)E * 4);
  int*    dstA   = (int*)alloc((size_t)E * 4);
  int*    counts = (int*)alloc((size_t)N * 4);
  float*  sbuf   = (float*)alloc((size_t)N * SDIM * 4);
  __bf16* sbf    = (__bf16*)alloc((size_t)N * SDIM * 2);
  float*  vbuf   = (float*)alloc((size_t)N * 3 * VDIM * 4);
  float*  rnb    = (float*)alloc((size_t)E * 3 * 4);
  uint32* geobf  = (uint32*)alloc((size_t)E * GEOW * 4);
  float*  ms_agg = (float*)alloc((size_t)N * SDIM * 4);
  float*  mv_agg = (float*)alloc((size_t)N * 3 * VDIM * 4);
  uint32* w1sw   = (uint32*)alloc((size_t)NLAYERS * KC1 * NT1 * 256 * 4);
  uint32* w2sw   = (uint32*)alloc((size_t)NLAYERS * KC2 * NT2 * 256 * 4);
  uint32* wlatsw = (uint32*)alloc((size_t)KC2 * 16 * 256 * 4);
  uint32* wn1sw  = (uint32*)alloc((size_t)KC2 * 16 * 256 * 4);
  uint32* wn2sw  = (uint32*)alloc((size_t)KC2 * 16 * 256 * 4);
  uint32* wg1sw  = (uint32*)alloc((size_t)KC2 * 16 * 256 * 4);
  __bf16* zbf    = (__bf16*)alloc((size_t)N * 256 * 2);
  __bf16* h1bf   = (__bf16*)alloc((size_t)N * 256 * 2);
  __bf16* hgbf   = (__bf16*)alloc((size_t)N * 256 * 2);
  float*  nodef  = (float*)alloc((size_t)N * 256 * 4);
  float*  gate   = (float*)alloc((size_t)N * 4);

  hipMemsetAsync(counts, 0, (size_t)N * 4, stream);
  hipMemsetAsync(vbuf,   0, (size_t)N * 3 * VDIM * 4, stream);

  k_center<<<(G + 255) / 256, 256, 0, stream>>>(pos, posc, G);
  k_copy_edges<<<(E + 255) / 256, 256, 0, stream>>>(eloc, eglo, EL, EG, srcA, dstA);
  k_embed<<<((size_t)N * SDIM + 255) / 256, 256, 0, stream>>>(x, W_atom, b_atom, sbuf, sbf, N);
  k_geom<<<(E + 255) / 256, 256, 0, stream>>>(srcA, dstA, posc, rnb, geobf, counts, E);

  // weight conversion + WMMA-B swizzle
  {
    long t1 = (long)NLAYERS * KC1 * NT1 * 256;
    k_swizzle<<<(t1 + 255) / 256, 256, 0, stream>>>(Wm1, w1sw, EDGE_IN, 256, KC1, NT1, NLAYERS);
    long t2 = (long)NLAYERS * KC2 * NT2 * 256;
    k_swizzle<<<(t2 + 255) / 256, 256, 0, stream>>>(Wm2, w2sw, 256, MSG_OUT, KC2, NT2, NLAYERS);
    long t3 = (long)KC2 * 16 * 256;
    k_swizzle<<<(t3 + 255) / 256, 256, 0, stream>>>(W_lat, wlatsw, 256, 256, KC2, 16, 1);
    k_swizzle<<<(t3 + 255) / 256, 256, 0, stream>>>(Wn1, wn1sw, 256, 256, KC2, 16, 1);
    k_swizzle<<<(t3 + 255) / 256, 256, 0, stream>>>(Wn2, wn2sw, 256, 256, KC2, 16, 1);
    k_swizzle<<<(t3 + 255) / 256, 256, 0, stream>>>(Wg1, wg1sw, 256, 256, KC2, 16, 1);
  }

  // message passing layers
  const int edgeBlocks = (E + EPB - 1) / EPB;
  const int nodeBlocks = (N + 7) / 8;
  for (int l = 0; l < NLAYERS; ++l) {
    hipMemsetAsync(ms_agg, 0, (size_t)N * SDIM * 4, stream);
    hipMemsetAsync(mv_agg, 0, (size_t)N * 3 * VDIM * 4, stream);
    k_edge_fused<<<edgeBlocks, 128, EDGE_SMEM, stream>>>(
        (const uint32*)sbf, geobf, srcA, dstA, rnb, vbuf,
        w1sw + (size_t)l * KC1 * NT1 * 256, bm1 + (size_t)l * SDIM,
        w2sw + (size_t)l * KC2 * NT2 * 256, bm2 + (size_t)l * MSG_OUT,
        ms_agg, mv_agg, E);
    k_node_update<<<nodeBlocks, 256, 0, stream>>>(
        sbuf, sbf, vbuf, ms_agg, mv_agg, counts,
        ln_g + (size_t)l * SDIM, ln_b + (size_t)l * SDIM, N);
  }

  // latent head: z, node features, gate path (all WMMA)
  const int gemmBlocks = (N + 15) / 16;
  k_gemm_node<<<gemmBlocks, 128, 0, stream>>>((const uint32*)sbf, wlatsw, b_lat,
                                              nullptr, zbf, 0, N);
  k_gemm_node<<<gemmBlocks, 128, 0, stream>>>((const uint32*)zbf, wn1sw, bn1,
                                              nullptr, h1bf, 1, N);
  k_gemm_node<<<gemmBlocks, 128, 0, stream>>>((const uint32*)h1bf, wn2sw, bn2,
                                              nodef, nullptr, 0, N);
  k_gemm_node<<<gemmBlocks, 128, 0, stream>>>((const uint32*)zbf, wg1sw, bg1,
                                              nullptr, hgbf, 1, N);
  k_gate<<<(N + 255) / 256, 256, 0, stream>>>(hgbf, Wg2, bg2, gate, N);
  k_pool<<<(G + 7) / 8, 256, 0, stream>>>(gate, nodef, (float*)d_out, G);
}
